// MSAPairWeightedAveraging_53300544143975
// MI455X (gfx1250) — matrix operand order repaired and checked by reading, not compiled, not run on hardware
//
#include <hip/hip_runtime.h>
#include <hip/hip_bf16.h>
#include <math.h>

typedef _Float16 v16h __attribute__((ext_vector_type(16)));
typedef _Float16 v8h  __attribute__((ext_vector_type(8)));
typedef float    v8f  __attribute__((ext_vector_type(8)));

#define S_DIM 512
#define N_DIM 384
#define CM    64
#define CZ    128
#define H_DIM 8
#define C_DIM 32
#define HC    256
#define LN_EPS 1e-5f

// ---------------------------------------------------------------------------
// Kernel 0: W4 [256][64] f32  ->  W4T [64][256] f16  (B-matrix friendly)
// ---------------------------------------------------------------------------
__global__ void prep_w4(const float* __restrict__ W4, _Float16* __restrict__ W4T) {
    int t = blockIdx.x * blockDim.x + threadIdx.x;   // 0..16383
    int d = t >> 8;          // output feature (column of W4)
    int k = t & 255;         // contraction index
    W4T[t] = (_Float16)W4[k * CM + d];
}

// ---------------------------------------------------------------------------
// Kernel 1: LayerNorm(m) then v = ln@W1 (f16, transposed for WMMA B),
//           g = sigmoid(ln@W3) (f16). One row (s,n) per block, 256 threads.
// vT layout: [S][H][C][N(j)]   gbuf layout: [S*N][256]
// ---------------------------------------------------------------------------
__global__ void ln_m_proj(const float* __restrict__ m,
                          const float* __restrict__ gamma,
                          const float* __restrict__ beta,
                          const float* __restrict__ W1,
                          const float* __restrict__ W3,
                          _Float16* __restrict__ vT,
                          _Float16* __restrict__ gbuf) {
    __shared__ float xs[CM];
    __shared__ float xn[CM];
    int row = blockIdx.x;                 // s*384 + n
    int s = row / N_DIM, n = row % N_DIM;
    int t = threadIdx.x;                  // 0..255
    if (t < CM) xs[t] = m[(size_t)row * CM + t];
    __syncthreads();
    float sum = 0.f, sq = 0.f;
#pragma unroll
    for (int k = 0; k < CM; ++k) { float x = xs[k]; sum += x; sq += x * x; }
    float mu  = sum * (1.f / CM);
    float var = sq * (1.f / CM) - mu * mu;
    float rs  = rsqrtf(var + LN_EPS);
    if (t < CM) xn[t] = (xs[t] - mu) * rs * gamma[t] + beta[t];
    __syncthreads();
    float a1 = 0.f, a3 = 0.f;
#pragma unroll 8
    for (int k = 0; k < CM; ++k) {
        float x = xn[k];
        a1 += x * W1[k * HC + t];
        a3 += x * W3[k * HC + t];
    }
    float gv = 1.f / (1.f + __expf(-a3));
    int h = t >> 5, c = t & 31;
    vT[(((size_t)s * H_DIM + h) * C_DIM + c) * N_DIM + n] = (_Float16)a1;
    gbuf[(size_t)row * HC + t] = (_Float16)gv;
}

// ---------------------------------------------------------------------------
// Kernel 2: b = LayerNorm(z) @ W2, stored as [H][j][i] (softmax axis i contig)
// One row (i,j) per block, 128 threads.
// ---------------------------------------------------------------------------
__global__ void ln_z_bias(const float* __restrict__ z,
                          const float* __restrict__ gz,
                          const float* __restrict__ bz,
                          const float* __restrict__ W2,
                          float* __restrict__ b_hji) {
    __shared__ float zs[CZ];
    __shared__ float zn[CZ];
    __shared__ float red[CZ];
    int row = blockIdx.x;                  // i*384 + j
    int i = row / N_DIM, j = row % N_DIM;
    int t = threadIdx.x;                   // 0..127
    zs[t] = z[(size_t)row * CZ + t];
    __syncthreads();
    float sum = 0.f, sq = 0.f;
#pragma unroll 8
    for (int k = 0; k < CZ; ++k) { float x = zs[k]; sum += x; sq += x * x; }
    float mu  = sum * (1.f / CZ);
    float var = sq * (1.f / CZ) - mu * mu;
    float rs  = rsqrtf(var + LN_EPS);
    zn[t] = (zs[t] - mu) * rs * gz[t] + bz[t];
    __syncthreads();
    int h = t & 7, grp = t >> 3;           // 16 k-groups per output h
    float p = 0.f;
#pragma unroll
    for (int mb = 0; mb < 8; ++mb) {
        int k = grp + 16 * mb;
        p += zn[k] * W2[k * H_DIM + h];
    }
    red[t] = p;
    __syncthreads();
    if (t < H_DIM) {
        float acc = 0.f;
#pragma unroll
        for (int g = 0; g < 16; ++g) acc += red[g * 8 + t];
        b_hji[((size_t)t * N_DIM + j) * N_DIM + i] = acc;
    }
}

// ---------------------------------------------------------------------------
// Kernel 3: softmax over i for each (h,j); write wT f16 [H][i][j] (A-matrix)
// Block = 384 threads (12 waves), one (h,j) per block.
// ---------------------------------------------------------------------------
__global__ void softmax_i(const float* __restrict__ b_hji,
                          _Float16* __restrict__ wT) {
    __shared__ float sred[12];
    int hj = blockIdx.x;
    int h = hj / N_DIM, j = hj % N_DIM;
    int i = threadIdx.x;                   // 0..383
    int wid  = threadIdx.x >> 5;
    int lane = threadIdx.x & 31;
    float v = b_hji[((size_t)h * N_DIM + j) * N_DIM + i];

    float mx = v;
    for (int off = 16; off > 0; off >>= 1) mx = fmaxf(mx, __shfl_xor(mx, off, 32));
    if (lane == 0) sred[wid] = mx;
    __syncthreads();
    if (wid == 0) {
        float tv = (lane < 12) ? sred[lane] : -1e30f;
        for (int off = 16; off > 0; off >>= 1) tv = fmaxf(tv, __shfl_xor(tv, off, 32));
        if (lane == 0) sred[0] = tv;
    }
    __syncthreads();
    float gmax = sred[0];
    __syncthreads();

    float e = __expf(v - gmax);
    float sm = e;
    for (int off = 16; off > 0; off >>= 1) sm += __shfl_xor(sm, off, 32);
    if (lane == 0) sred[wid] = sm;
    __syncthreads();
    if (wid == 0) {
        float tv = (lane < 12) ? sred[lane] : 0.f;
        for (int off = 16; off > 0; off >>= 1) tv += __shfl_xor(tv, off, 32);
        if (lane == 0) sred[0] = tv;
    }
    __syncthreads();
    float w = e / sred[0];
    wT[((size_t)h * N_DIM + i) * N_DIM + j] = (_Float16)w;
}

// ---------------------------------------------------------------------------
// Kernel 4: O[s,i,h,c] = sum_j w[h,i,j] * v[s,h,c,j]; then gate and store f16.
// One block per (h,s); 192 threads = 6 waves. Each wave owns 64 i-rows and
// the full c=32 => 4 i-tiles x 2 c-tiles = 8 independent 16x16 accumulators.
// K loop: 12 steps x 8 WMMA; 12 b128 loads per step (1.5 loads / WMMA).
// Fragment layout used here (A and B share one k order):
//   A (16x32 f16): lane M = lane&15; half=lane>>4 -> K 0..15 / 16..31 (contiguous)
//   B (32x16 f16): lane N = lane&15; half -> K 0..15 / 16..31 (contiguous)
//   C (16x16 f32): elem r -> M = half*8+r, N = lane&15
// ---------------------------------------------------------------------------
__global__ void pwa_einsum(const _Float16* __restrict__ wT,   // [H][i][j]
                           const _Float16* __restrict__ vT,   // [S][H][c][j]
                           const _Float16* __restrict__ gbuf, // [S*N][256]
                           _Float16* __restrict__ obuf) {     // [S*N][256]
    int h = blockIdx.x;
    int s = blockIdx.y;
    int wave = threadIdx.x >> 5;           // 0..5
    int lane = threadIdx.x & 31;
    int half = lane >> 4, lcol = lane & 15;
    int ibase = wave * 64;

    const _Float16* Ab[4];
    const _Float16* Bb[2];
#pragma unroll
    for (int t = 0; t < 4; ++t)
        Ab[t] = wT + ((size_t)h * N_DIM + (ibase + t * 16 + lcol)) * N_DIM + half * 16;
#pragma unroll
    for (int ct = 0; ct < 2; ++ct)
        Bb[ct] = vT + (((size_t)s * H_DIM + h) * C_DIM + ct * 16 + lcol) * N_DIM + half * 16;

    v8f acc[4][2];
#pragma unroll
    for (int t = 0; t < 4; ++t)
#pragma unroll
        for (int ct = 0; ct < 2; ++ct)
            acc[t][ct] = (v8f){0.f, 0.f, 0.f, 0.f, 0.f, 0.f, 0.f, 0.f};

#pragma unroll 2
    for (int j0 = 0; j0 < N_DIM; j0 += 32) {
        // prefetch next K-block (emits global_prefetch_b8)
        if (j0 + 32 < N_DIM) {
            __builtin_prefetch(Bb[0] + j0 + 32, 0, 1);
            __builtin_prefetch(Ab[0] + j0 + 32, 0, 1);
        }
        v16h b[2];
#pragma unroll
        for (int ct = 0; ct < 2; ++ct)
            b[ct] = *(const v16h*)(Bb[ct] + j0);
        v16h a[4];
#pragma unroll
        for (int t = 0; t < 4; ++t) {
            v8h a0 = *(const v8h*)(Ab[t] + j0);
            v8h a1 = *(const v8h*)(Ab[t] + j0 + 8);
            a[t] = __builtin_shufflevector(a0, a1,
                                           0,1,2,3,4,5,6,7,8,9,10,11,12,13,14,15);
        }
#pragma unroll
        for (int t = 0; t < 4; ++t)
#pragma unroll
            for (int ct = 0; ct < 2; ++ct)
                acc[t][ct] = __builtin_amdgcn_wmma_f32_16x16x32_f16(
                    false, a[t], false, b[ct], (short)0, acc[t][ct], false, false);
    }

    // gate + store f16
#pragma unroll
    for (int t = 0; t < 4; ++t) {
#pragma unroll
        for (int ct = 0; ct < 2; ++ct) {
            int cc = ct * 16 + lcol;
#pragma unroll
            for (int r = 0; r < 8; ++r) {
                int i = ibase + t * 16 + half * 8 + r;
                size_t idx = ((size_t)s * N_DIM + i) * HC + h * C_DIM + cc;
                float gv = (float)gbuf[idx];
                obuf[idx] = (_Float16)(acc[t][ct][r] * gv);
            }
        }
    }
}

// ---------------------------------------------------------------------------
// Kernel 5: out = o @ W4 via WMMA. Rows = S*N, K = 256, N = 64.
// Block = 256 threads = 8 waves covering 32 rows x 64 cols (each wave 16x16).
// ---------------------------------------------------------------------------
__global__ void out_proj(const _Float16* __restrict__ obuf, // [R][256]
                         const _Float16* __restrict__ W4T,  // [64][256]
                         float* __restrict__ out) {         // [R][64]
    int wave = threadIdx.x >> 5;
    int lane = threadIdx.x & 31;
    int half = lane >> 4, lcol = lane & 15;
    int tile_m = wave >> 2, tile_n = wave & 3;
    int row0 = blockIdx.x * 32 + tile_m * 16;
    int n0   = tile_n * 16;

    const _Float16* Abase = obuf + ((size_t)(row0 + lcol)) * HC + half * 16;
    const _Float16* Bbase = W4T  + ((size_t)(n0 + lcol)) * HC + half * 16;

    v8f acc = {0.f, 0.f, 0.f, 0.f, 0.f, 0.f, 0.f, 0.f};
#pragma unroll
    for (int k0 = 0; k0 < HC; k0 += 32) {
        v8h a0 = *(const v8h*)(Abase + k0);
        v8h a1 = *(const v8h*)(Abase + k0 + 8);
        v16h a = __builtin_shufflevector(a0, a1, 0,1,2,3,4,5,6,7,8,9,10,11,12,13,14,15);
        v16h b = *(const v16h*)(Bbase + k0);
        acc = __builtin_amdgcn_wmma_f32_16x16x32_f16(false, a, false, b,
                                                     (short)0, acc, false, false);
    }
#pragma unroll
    for (int r = 0; r < 8; ++r) {
        out[((size_t)(row0 + half * 8 + r)) * CM + n0 + lcol] = acc[r];
    }
}

// ---------------------------------------------------------------------------
// Launch
// ---------------------------------------------------------------------------
extern "C" void kernel_launch(void* const* d_in, const int* in_sizes, int n_in,
                              void* d_out, int out_size, void* d_ws, size_t ws_size,
                              hipStream_t stream) {
    const float* m_si = (const float*)d_in[0];
    const float* z_ij = (const float*)d_in[1];
    const float* g_m  = (const float*)d_in[2];
    const float* b_m  = (const float*)d_in[3];
    const float* W1   = (const float*)d_in[4];
    const float* g_z  = (const float*)d_in[5];
    const float* b_z  = (const float*)d_in[6];
    const float* W2   = (const float*)d_in[7];
    const float* W3   = (const float*)d_in[8];
    const float* W4   = (const float*)d_in[9];
    float* out = (float*)d_out;

    // Workspace layout (bytes):
    char* ws = (char*)d_ws;
    _Float16* vT   = (_Float16*)(ws);                       // 100,663,296 B
    _Float16* gbuf = (_Float16*)(ws + 100663296ull);        // 100,663,296 B
    _Float16* obuf = (_Float16*)(ws + 201326592ull);        // 100,663,296 B
    float*    b_hji= (float*)   (ws + 301989888ull);        //   4,718,592 B
    _Float16* wT   = (_Float16*)(ws + 306708480ull);        //   2,359,296 B
    _Float16* W4T  = (_Float16*)(ws + 309067776ull);        //      32,768 B

    prep_w4  <<<64, 256, 0, stream>>>(W4, W4T);
    ln_m_proj<<<S_DIM * N_DIM, 256, 0, stream>>>(m_si, g_m, b_m, W1, W3, vT, gbuf);
    ln_z_bias<<<N_DIM * N_DIM, 128, 0, stream>>>(z_ij, g_z, b_z, W2, b_hji);
    softmax_i<<<H_DIM * N_DIM, 384, 0, stream>>>(b_hji, wT);

    dim3 g4(H_DIM, S_DIM);
    pwa_einsum<<<g4, 192, 0, stream>>>(wT, vT, gbuf, obuf);

    out_proj<<<(S_DIM * N_DIM) / 32, 256, 0, stream>>>(obuf, W4T, out);
}
